// MultiHeadedAttentionWithRelations_9835475108431
// MI455X (gfx1250) — matrix operand order repaired and checked by reading, not compiled, not run on hardware
//
#include <hip/hip_runtime.h>
#include <math.h>

// Problem constants (from reference setup_inputs)
#define TB 4      // batch
#define TS 512    // sequence
#define TD 1024   // model dim
#define THH 8     // heads
#define TDK 128   // head dim

typedef __attribute__((ext_vector_type(16))) __bf16 bf16x16;
typedef __attribute__((ext_vector_type(2)))  __bf16 bf16x2;
typedef __attribute__((ext_vector_type(8)))  float   f32x8;
typedef __attribute__((ext_vector_type(8)))  int     i32x8;

#define WMMA_BF16(a, b, c) \
  __builtin_amdgcn_wmma_f32_16x16x32_bf16(false, (a), false, (b), (short)0, (c), false, false)

#if defined(__has_builtin)
#if __has_builtin(__builtin_amdgcn_cvt_pk_bf16_f32)
#define HAVE_CVT_PK_BF16 1
#endif
#endif

// ---- bf16 helpers ----
// Pack two f32 -> one dword of two bf16 (RNE).
static __device__ __forceinline__ unsigned pk2(float lo, float hi) {
#ifdef HAVE_CVT_PK_BF16
  bf16x2 v = __builtin_amdgcn_cvt_pk_bf16_f32(lo, hi);
  return __builtin_bit_cast(unsigned, v);
#else
  unsigned ua = __builtin_bit_cast(unsigned, lo);
  unsigned ub = __builtin_bit_cast(unsigned, hi);
  unsigned ra = ua + 0x7FFFu + ((ua >> 16) & 1u);
  unsigned rb = ub + 0x7FFFu + ((ub >> 16) & 1u);
  // result = { rb[31:16], ra[31:16] }
  return __builtin_amdgcn_perm(rb, ra, 0x07060302u);
#endif
}
static __device__ __forceinline__ __bf16 f2bf(float f) {
  unsigned u = __builtin_bit_cast(unsigned, f);
  unsigned short s = (unsigned short)((u + 0x7FFFu + ((u >> 16) & 1u)) >> 16);
  return __builtin_bit_cast(__bf16, s);
}
static __device__ __forceinline__ bf16x16 zero_bf() {
  i32x8 z = {0, 0, 0, 0, 0, 0, 0, 0};
  return __builtin_bit_cast(bf16x16, z);
}

// ---- WMMA fragment loaders (contiguous k order, same for A and B) ----
// A 16x32 bf16: lane&15 = M row; lanes 0-15 hold K {0..15}, lanes 16-31 hold {16..31}
static __device__ __forceinline__ bf16x16 ldA_bf16(const __bf16* __restrict__ row, int kb, int lane) {
  const int half = (lane >> 4) & 1;
  const unsigned* p0 = (const unsigned*)(row + kb + half * 16);
  i32x8 r;
#pragma unroll
  for (int j = 0; j < 8; ++j) r[j] = p0[j];
  return __builtin_bit_cast(bf16x16, r);
}
// B 32x16 bf16 where memory holds Bsrc[n][k] row-major (K contiguous): nrow = &Bsrc[n*ld]
static __device__ __forceinline__ bf16x16 ldB_nk_bf16(const __bf16* __restrict__ nrow, int kb, int lane) {
  const unsigned* p = (const unsigned*)(nrow + kb + ((lane >> 4) & 1) * 16);
  i32x8 r;
#pragma unroll
  for (int j = 0; j < 8; ++j) r[j] = p[j];
  return __builtin_bit_cast(bf16x16, r);
}
static __device__ __forceinline__ bf16x16 ldB_nk_f32(const float* __restrict__ nrow, int kb, int lane) {
  const float* p = nrow + kb + ((lane >> 4) & 1) * 16;
  i32x8 r;
#pragma unroll
  for (int j = 0; j < 8; ++j) r[j] = pk2(p[2 * j], p[2 * j + 1]);
  return __builtin_bit_cast(bf16x16, r);
}
// B 32x16 where memory holds Bsrc[k][n] (K x N row-major, stride ld): strided gather + convert
static __device__ __forceinline__ bf16x16 ldB_kn_f32(const float* __restrict__ base, int kb,
                                                     int ncol, int ld, int lane) {
  const int ko = kb + ((lane >> 4) & 1) * 16;
  i32x8 r;
#pragma unroll
  for (int j = 0; j < 8; ++j)
    r[j] = pk2(base[(size_t)(ko + 2 * j) * ld + ncol],
               base[(size_t)(ko + 2 * j + 1) * ld + ncol]);
  return __builtin_bit_cast(bf16x16, r);
}

// ============ Kernel 0: bulk f32 -> bf16 conversion (one pass, vectorized) ============
__global__ __launch_bounds__(256)
void cvt_bf16_kernel(const float* __restrict__ in, __bf16* __restrict__ out, int n) {
  const int i = (blockIdx.x * 256 + threadIdx.x) * 8;
  if (i >= n) return;
  const float4* src = (const float4*)(in + i);
  float4 a = src[0], b = src[1];
  uint4 o;
  o.x = pk2(a.x, a.y);
  o.y = pk2(a.z, a.w);
  o.z = pk2(b.x, b.y);
  o.w = pk2(b.z, b.w);
  *(uint4*)(out + i) = o;
}

// ==== Kernel 1: C = A @ W^T + bias (torch Linear), pure-bf16 inner loop ====
// mode 0: f32 row-major [M,N] -> outF
// mode 1: bf16 head-split [B,H,S,dk] -> outB
// mode 2: bf16 head-split dk-transposed [B,H,dk,S] -> outB
__global__ __launch_bounds__(256)
void gemm_bias_kernel(const __bf16* __restrict__ A, const __bf16* __restrict__ W,
                      const float* __restrict__ bias, float* __restrict__ outF,
                      __bf16* __restrict__ outB, int M, int N, int K, int mode) {
  const int lane = threadIdx.x & 31, wid = threadIdx.x >> 5;
  const int wm = wid & 3, wn = wid >> 2;
  const int m0 = blockIdx.y * 64 + wm * 16;
  const int n0 = blockIdx.x * 128 + wn * 64;
  const __bf16* Arow = A + (size_t)(m0 + (lane & 15)) * K;
  f32x8 z = {0, 0, 0, 0, 0, 0, 0, 0};
  f32x8 acc[4];
#pragma unroll
  for (int t = 0; t < 4; ++t) acc[t] = z;
  for (int kb = 0; kb < K; kb += 32) {
    bf16x16 a = ldA_bf16(Arow, kb, lane);
#pragma unroll
    for (int t = 0; t < 4; ++t) {
      const __bf16* Wrow = W + (size_t)(n0 + t * 16 + (lane & 15)) * K;
      bf16x16 b = ldB_nk_bf16(Wrow, kb, lane);
      acc[t] = WMMA_BF16(a, b, acc[t]);
    }
  }
  const int roff = (lane < 16) ? 0 : 8;
#pragma unroll
  for (int t = 0; t < 4; ++t) {
    const int n = n0 + t * 16 + (lane & 15);
    const float bv = bias[n];
#pragma unroll
    for (int r = 0; r < 8; ++r) {
      const int m = m0 + r + roff;
      const float v = acc[t][r] + bv;
      if (mode == 0) {
        outF[(size_t)m * N + n] = v;
      } else {
        const int b_ = m / TS, s_ = m % TS, h_ = n / TDK, d_ = n % TDK;
        const size_t idx = (mode == 1)
            ? ((((size_t)b_ * THH + h_) * TS + s_) * TDK + d_)
            : ((((size_t)b_ * THH + h_) * TDK + d_) * TS + s_);
        outB[idx] = f2bf(v);
      }
    }
  }
}

// ===== Kernel 2: rel_s[b,h,q,k] = sum_d q[b,h,q,d]*relation_k[b,q,k,d]; block per (b,qpos) =====
// A = q for all 8 heads (M rows 0..7, rows 8..15 zero); B = relation_k[b,qpos] (contiguous K=d)
__global__ __launch_bounds__(256)
void rel_scores_kernel(const __bf16* __restrict__ qb, const float* __restrict__ relk,
                       float* __restrict__ rels) {
  const int lane = threadIdx.x & 31, wid = threadIdx.x >> 5;
  const int bq = blockIdx.x;  // b*S + qpos
  const int b = bq / TS, qpos = bq % TS;
  const int r = lane & 15;
  bf16x16 afr[4];
#pragma unroll
  for (int ki = 0; ki < 4; ++ki) {
    if (r < THH) {
      const __bf16* qrow = qb + (((size_t)b * THH + r) * TS + qpos) * TDK;
      afr[ki] = ldA_bf16(qrow, ki * 32, lane);
    } else {
      afr[ki] = zero_bf();
    }
  }
  const float* rkbase = relk + (size_t)bq * TS * TDK;  // [kpos][d]
  f32x8 z = {0, 0, 0, 0, 0, 0, 0, 0};
#pragma unroll
  for (int t = 0; t < 4; ++t) {
    const int ntile = wid * 64 + t * 16;  // kpos tile
    f32x8 acc = z;
#pragma unroll
    for (int ki = 0; ki < 4; ++ki) {
      const float* nrow = rkbase + (size_t)(ntile + r) * TDK;
      bf16x16 bfr = ldB_nk_f32(nrow, ki * 32, lane);
      acc = WMMA_BF16(afr[ki], bfr, acc);
    }
    if (lane < 16) {
#pragma unroll
      for (int h = 0; h < THH; ++h)
        rels[(((size_t)b * THH + h) * TS + qpos) * TS + ntile + lane] = acc[h];
    }
  }
}

// ============ Kernel 3: scores = q @ k^T per (b,h); 64(q) x 128(k) tile / block ============
__global__ __launch_bounds__(256)
void qk_kernel(const __bf16* __restrict__ qb, const __bf16* __restrict__ kbm,
               float* __restrict__ scores) {
  const int lane = threadIdx.x & 31, wid = threadIdx.x >> 5;
  const int wm = wid & 3, wn = wid >> 2;
  const int bh = blockIdx.z;
  const int m0 = blockIdx.y * 64 + wm * 16;   // qpos
  const int n0 = blockIdx.x * 128 + wn * 64;  // kpos
  const __bf16* qrow = qb + ((size_t)bh * TS + m0 + (lane & 15)) * TDK;
  f32x8 z = {0, 0, 0, 0, 0, 0, 0, 0};
  f32x8 acc[4];
#pragma unroll
  for (int t = 0; t < 4; ++t) acc[t] = z;
  for (int kd = 0; kd < TDK; kd += 32) {
    bf16x16 a = ldA_bf16(qrow, kd, lane);
#pragma unroll
    for (int t = 0; t < 4; ++t) {
      const __bf16* krow = kbm + ((size_t)bh * TS + n0 + t * 16 + (lane & 15)) * TDK;
      bf16x16 b = ldB_nk_bf16(krow, kd, lane);
      acc[t] = WMMA_BF16(a, b, acc[t]);
    }
  }
  const int roff = (lane < 16) ? 0 : 8;
#pragma unroll
  for (int t = 0; t < 4; ++t) {
    const int n = n0 + t * 16 + (lane & 15);
#pragma unroll
    for (int r = 0; r < 8; ++r) {
      const int m = m0 + r + roff;
      scores[((size_t)bh * TS + m) * TS + n] = acc[t][r];
    }
  }
}

// ===== Kernel 4: softmax over (qk + rel)/sqrt(dk) with mask; one row per block =====
__global__ __launch_bounds__(256)
void softmax_kernel(const float* __restrict__ scores, const float* __restrict__ rels,
                    const int* __restrict__ mask, __bf16* __restrict__ p) {
  const int row = blockIdx.x;  // (b*H + h)*S + qpos
  const int qpos = row % TS;
  const int b = row / (THH * TS);
  const int tid = threadIdx.x;
  const float* sc = scores + (size_t)row * TS;
  const float* rl = rels + (size_t)row * TS;
  const int* mk = mask + ((size_t)b * TS + qpos) * TS;
  __shared__ float red[256];
  const float scale = 0.0883883476483184f;  // 1/sqrt(128)
  float v[2];
  float mx = -3.0e38f;
#pragma unroll
  for (int i = 0; i < 2; ++i) {
    const int k = tid + i * 256;
    float x = (sc[k] + rl[k]) * scale;
    if (mk[k] == 0) x = -1.0e9f;
    v[i] = x;
    mx = fmaxf(mx, x);
  }
  red[tid] = mx; __syncthreads();
  for (int s = 128; s > 0; s >>= 1) {
    if (tid < s) red[tid] = fmaxf(red[tid], red[tid + s]);
    __syncthreads();
  }
  mx = red[0]; __syncthreads();
  float sum = 0.f;
#pragma unroll
  for (int i = 0; i < 2; ++i) { v[i] = __expf(v[i] - mx); sum += v[i]; }
  red[tid] = sum; __syncthreads();
  for (int s = 128; s > 0; s >>= 1) {
    if (tid < s) red[tid] += red[tid + s];
    __syncthreads();
  }
  const float inv = 1.0f / red[0];
#pragma unroll
  for (int i = 0; i < 2; ++i) p[(size_t)row * TS + tid + i * 256] = f2bf(v[i] * inv);
}

// ===== Kernel 5: attn = p @ v per (b,h); v stored transposed [B,H,dk,S] -> contiguous B loads ====
__global__ __launch_bounds__(256)
void pv_kernel(const __bf16* __restrict__ p, const __bf16* __restrict__ vT,
               float* __restrict__ attn) {
  const int lane = threadIdx.x & 31, wid = threadIdx.x >> 5;
  const int wm = wid & 3, wn = wid >> 2;
  const int bh = blockIdx.y;
  const int b = bh / THH, h = bh % THH;
  const int m0 = blockIdx.x * 64 + wm * 16;
  const int n0 = wn * 64;
  const __bf16* prow = p + ((size_t)bh * TS + m0 + (lane & 15)) * TS;
  const __bf16* vbase = vT + (size_t)bh * TDK * TS;  // [d][kpos]
  f32x8 z = {0, 0, 0, 0, 0, 0, 0, 0};
  f32x8 acc[4];
#pragma unroll
  for (int t = 0; t < 4; ++t) acc[t] = z;
  for (int kp = 0; kp < TS; kp += 32) {
    bf16x16 a = ldA_bf16(prow, kp, lane);
#pragma unroll
    for (int t = 0; t < 4; ++t) {
      const __bf16* nrow = vbase + (size_t)(n0 + t * 16 + (lane & 15)) * TS;
      bf16x16 b2 = ldB_nk_bf16(nrow, kp, lane);
      acc[t] = WMMA_BF16(a, b2, acc[t]);
    }
  }
  const int roff = (lane < 16) ? 0 : 8;
#pragma unroll
  for (int t = 0; t < 4; ++t) {
    const int d = n0 + t * 16 + (lane & 15);
#pragma unroll
    for (int r = 0; r < 8; ++r) {
      const int m = m0 + r + roff;
      attn[((size_t)b * TS + m) * TD + h * TDK + d] = acc[t][r];
    }
  }
}

// ===== Kernel 6: attn += p . relation_v, per (b,qpos) across all heads (M=8 padded to 16) =====
__global__ __launch_bounds__(256)
void prv_kernel(const __bf16* __restrict__ p, const float* __restrict__ relv,
                float* __restrict__ attn) {
  const int lane = threadIdx.x & 31, wid = threadIdx.x >> 5;
  const int bq = blockIdx.x;  // b*S + qpos
  const int b = bq / TS, qpos = bq % TS;
  const int r = lane & 15;
  const int ntile = wid * 16;  // d tile (8 waves * 16 = 128 = dk)
  const float* rvbase = relv + (size_t)bq * TS * TDK;  // [kpos][d]
  f32x8 acc = {0, 0, 0, 0, 0, 0, 0, 0};
  for (int kp = 0; kp < TS; kp += 32) {
    bf16x16 a;
    if (r < THH) {
      const __bf16* prow = p + (((size_t)b * THH + r) * TS + qpos) * TS;
      a = ldA_bf16(prow, kp, lane);
    } else {
      a = zero_bf();
    }
    bf16x16 b2 = ldB_kn_f32(rvbase, kp, ntile + r, TDK, lane);
    acc = WMMA_BF16(a, b2, acc);
  }
  if (lane < 16) {
    float* dst = attn + ((size_t)b * TS + qpos) * TD;
#pragma unroll
    for (int h = 0; h < THH; ++h) dst[h * TDK + ntile + lane] += acc[h];
  }
}

extern "C" void kernel_launch(void* const* d_in, const int* in_sizes, int n_in,
                              void* d_out, int out_size, void* d_ws, size_t ws_size,
                              hipStream_t stream) {
  (void)in_sizes; (void)n_in; (void)out_size; (void)ws_size;
  const float* query = (const float*)d_in[0];
  const float* key   = (const float*)d_in[1];
  const float* value = (const float*)d_in[2];
  const float* relk  = (const float*)d_in[3];
  const float* relv  = (const float*)d_in[4];
  const int*   mask  = (const int*)d_in[5];
  const float* wq = (const float*)d_in[6];  const float* bq = (const float*)d_in[7];
  const float* wk = (const float*)d_in[8];  const float* bk = (const float*)d_in[9];
  const float* wv = (const float*)d_in[10]; const float* bv = (const float*)d_in[11];
  const float* wo = (const float*)d_in[12]; const float* bo = (const float*)d_in[13];
  float* out = (float*)d_out;

  // Workspace carve-out (~106 MB total)
  char* w = (char*)d_ws;
  const size_t QKV_B = (size_t)TB * THH * TS * TDK * 2;  // bf16 [B,H,S,dk]
  const size_t SC_B  = (size_t)TB * THH * TS * TS * 4;   // f32  [B,H,S,S]
  const size_t P_B   = (size_t)TB * THH * TS * TS * 2;   // bf16 [B,H,S,S]
  __bf16* q_bf = (__bf16*)w; w += QKV_B;
  __bf16* k_bf = (__bf16*)w; w += QKV_B;
  __bf16* vT_bf = (__bf16*)w; w += QKV_B;
  float*  sc   = (float*)w;  w += SC_B;
  float*  rl   = (float*)w;  w += SC_B;
  __bf16* p_bf = (__bf16*)w; w += P_B;
  float*  attn = (float*)w;  w += (size_t)TB * TS * TD * 4;  // f32 [B,S,D]
  __bf16* x_bf = (__bf16*)w; w += (size_t)TB * TS * TD * 2;  // staging: activations bf16
  __bf16* w_bf = (__bf16*)w;                                 // staging: weights bf16

  const int M = TB * TS, N = TD, K = TD;
  const int NA = M * K;  // activation elems (2M)
  const int NW = N * K;  // weight elems (1M)
  dim3 gGemm(N / 128, M / 64);
  const int gCvtA = NA / (256 * 8), gCvtW = NW / (256 * 8);

  // 1) q/k/v projections (convert operands to bf16, then pure-bf16 WMMA GEMM)
  cvt_bf16_kernel<<<gCvtA, 256, 0, stream>>>(query, x_bf, NA);
  cvt_bf16_kernel<<<gCvtW, 256, 0, stream>>>(wq, w_bf, NW);
  gemm_bias_kernel<<<gGemm, 256, 0, stream>>>(x_bf, w_bf, bq, nullptr, q_bf, M, N, K, 1);
  cvt_bf16_kernel<<<gCvtA, 256, 0, stream>>>(key, x_bf, NA);
  cvt_bf16_kernel<<<gCvtW, 256, 0, stream>>>(wk, w_bf, NW);
  gemm_bias_kernel<<<gGemm, 256, 0, stream>>>(x_bf, w_bf, bk, nullptr, k_bf, M, N, K, 1);
  cvt_bf16_kernel<<<gCvtA, 256, 0, stream>>>(value, x_bf, NA);
  cvt_bf16_kernel<<<gCvtW, 256, 0, stream>>>(wv, w_bf, NW);
  gemm_bias_kernel<<<gGemm, 256, 0, stream>>>(x_bf, w_bf, bv, nullptr, vT_bf, M, N, K, 2);
  // 2) relation scores (one pass over 512MB relation_k, all heads at once)
  rel_scores_kernel<<<TB * TS, 256, 0, stream>>>(q_bf, relk, rl);
  // 3) q @ k^T
  qk_kernel<<<dim3(TS / 128, TS / 64, TB * THH), 256, 0, stream>>>(q_bf, k_bf, sc);
  // 4) masked scaled softmax -> bf16 probs
  softmax_kernel<<<TB * THH * TS, 256, 0, stream>>>(sc, rl, mask, p_bf);
  // 5) p @ v (transposed v -> contiguous B-fragment loads)
  pv_kernel<<<dim3(TS / 64, TB * THH), 256, 0, stream>>>(p_bf, vT_bf, attn);
  // 6) attn += p . relation_v (one pass over 512MB relation_v)
  prv_kernel<<<TB * TS, 256, 0, stream>>>(p_bf, relv, attn);
  // 7) output projection -> d_out (f32)
  cvt_bf16_kernel<<<gCvtA, 256, 0, stream>>>(attn, x_bf, NA);
  cvt_bf16_kernel<<<gCvtW, 256, 0, stream>>>(wo, w_bf, NW);
  gemm_bias_kernel<<<gGemm, 256, 0, stream>>>(x_bf, w_bf, bo, out, nullptr, M, N, K, 0);
}
